// SC_ConvCapsuleLayer_31258771980609
// MI455X (gfx1250) — matrix operand order repaired and checked by reading, not compiled, not run on hardware
//
#include <hip/hip_runtime.h>

typedef __attribute__((ext_vector_type(16))) _Float16 v16h;
typedef __attribute__((ext_vector_type(8)))  _Float16 v8h;
typedef __attribute__((ext_vector_type(8)))  float    v8f;

#define NQ 26   // 25 kernel taps padded to even

// ---------------------------------------------------------------------------
// Pack conv input: x (B,16,4,64,64) f32  ->  inp[n][y][x][c] f16  (NHWC)
// IN[n,c,y,x] from the raw reinterpret: l = c*4096+y*64+x,
// h=l>>10, w=(l>>4)&63, a=l&15, b=n&7, ci=n>>3, value = x[b,a,ci,h,w].
// ---------------------------------------------------------------------------
__global__ void pack_input_kernel(const float* __restrict__ x,
                                  _Float16* __restrict__ inp) {
    int id = blockIdx.x * blockDim.x + threadIdx.x;   // exact: 32*64*64*16
    int c  = id & 15;
    int xw = (id >> 4) & 63;
    int yy = (id >> 10) & 63;
    int n  = id >> 16;
    int l  = (c << 12) | (yy << 6) | xw;
    int h  = l >> 10;
    int w  = (l >> 4) & 63;
    int a  = l & 15;
    int b  = n & 7;
    int ci = n >> 3;
    float val = x[((((b << 4) + a) << 2) + ci) * 4096 + (h << 6) + w];
    inp[id] = (_Float16)val;
}

// ---------------------------------------------------------------------------
// Pack weights: W (256,16,5,5) f32 -> wp[q][o][c] f16, q in [0,26), q==25 zero
// ---------------------------------------------------------------------------
__global__ void pack_w_kernel(const float* __restrict__ W,
                              _Float16* __restrict__ wp) {
    int id = blockIdx.x * blockDim.x + threadIdx.x;   // exact: 26*256*16
    int c = id & 15;
    int o = (id >> 4) & 255;
    int q = id >> 12;
    float val = 0.0f;
    if (q < 25) {
        int ky = q / 5, kx = q % 5;
        val = W[((o * 16 + c) * 5 + ky) * 5 + kx];
    }
    wp[id] = (_Float16)val;
}

// ---------------------------------------------------------------------------
// Implicit-GEMM conv via WMMA with LDS-staged B.
// Workgroup (8 waves) tile: M = 256 (all out-channels) x N = 64 (one image
// row).  Per K-chunk (2 taps x 16ch = 32): the 4KB B slice is staged once in
// LDS (single predicated b128 load per thread) and shared by all waves;
// A fragments are wave-disjoint b128 L2 loads.
// Each wave: 2 M-subtiles x 4 N-subtiles = 8 WMMAs per chunk, 64 acc VGPRs.
// ---------------------------------------------------------------------------
__global__ void __launch_bounds__(256)
conv_wmma_kernel(const _Float16* __restrict__ inp,
                 const _Float16* __restrict__ wp,
                 float* __restrict__ votes) {
    __shared__ _Float16 Bs[2][64][16];   // [tap][x][ch] = 4 KB

    int tid  = threadIdx.x;
    int wv   = tid >> 5;            // wave 0..7 -> out-channel group
    int lane = tid & 31;
    int hi   = lane >> 4;
    int lo   = lane & 15;

    int n = blockIdx.x >> 6;        // conv sample 0..31
    int y = blockIdx.x & 63;        // image row

    // staging role: 128 rows (2 taps x 64 x-positions) x 2 half-loads of 16B
    int srow  = tid >> 1;
    int shalf = tid & 1;
    int stap  = srow >> 6;
    int sx    = srow & 63;

    int o0 = (wv << 5) + lo;        // A-matrix row of M-subtile 0 (subtile 1: +16)

    v8f acc[2][4] = {};

    for (int q0 = 0; q0 < NQ; q0 += 2) {
        // ---- B slice: ONE predicated 16B vector load per thread ----
        int q  = q0 + stap;
        int iy = y  + (q / 5) - 2;
        int ix = sx + (q % 5) - 2;
        bool valid = (q < 25) && ((unsigned)iy < 64u) && ((unsigned)ix < 64u);
        v8h stgv = {};
        if (valid) {
            const v8h* src = reinterpret_cast<const v8h*>(
                inp + ((((((n << 6) + iy) << 6) + ix) << 4) + (shalf << 3)));
            stgv = *src;            // global_load_b128
        }

        __syncthreads();            // previous chunk's LDS reads complete
        *reinterpret_cast<v8h*>(&Bs[stap][sx][shalf << 3]) = stgv;  // ds_store_b128
        __syncthreads();

        // ---- A fragments: four b128 loads, packed per ISA layout ----
        // e<8 -> Wp[q0][o][hi*8+e], e>=8 -> Wp[q0+1][o][hi*8+(e-8)]
        const v8h* wa0 = reinterpret_cast<const v8h*>(
                             wp + ((((q0    ) << 8) + o0) << 4) + (hi << 3));
        const v8h* wa1 = reinterpret_cast<const v8h*>(
                             wp + ((((q0 + 1) << 8) + o0) << 4) + (hi << 3));
        v8h a0lo = wa0[0];          // M-subtile 0, tap q0
        v8h a0hi = wa1[0];          // M-subtile 0, tap q0+1
        v8h a1lo = wa0[32];         // M-subtile 1 (o0+16): +16*16 halves = 32 v8h
        v8h a1hi = wa1[32];
        v16h a0 = __builtin_shufflevector(a0lo, a0hi,
                      0,1,2,3,4,5,6,7,8,9,10,11,12,13,14,15);
        v16h a1 = __builtin_shufflevector(a1lo, a1hi,
                      0,1,2,3,4,5,6,7,8,9,10,11,12,13,14,15);

        // ---- 4 N-subtiles from LDS, 8 WMMAs ----
        #pragma unroll
        for (int nt = 0; nt < 4; ++nt) {
            v8h blo = *reinterpret_cast<const v8h*>(&Bs[0][(nt << 4) + lo][hi << 3]);
            v8h bhi = *reinterpret_cast<const v8h*>(&Bs[1][(nt << 4) + lo][hi << 3]);
            v16h bf = __builtin_shufflevector(blo, bhi,
                          0,1,2,3,4,5,6,7,8,9,10,11,12,13,14,15);
            acc[0][nt] = __builtin_amdgcn_wmma_f32_16x16x32_f16(
                             false, a0, false, bf, (short)0, acc[0][nt], false, false);
            acc[1][nt] = __builtin_amdgcn_wmma_f32_16x16x32_f16(
                             false, a1, false, bf, (short)0, acc[1][nt], false, false);
        }
    }

    // ---- store D: lane holds M = r + hi*8, N = lo ----
    int spb = (y << 6);
    #pragma unroll
    for (int m = 0; m < 2; ++m)
        #pragma unroll
        for (int nt = 0; nt < 4; ++nt)
            #pragma unroll
            for (int r = 0; r < 8; ++r) {
                int oo = (wv << 5) + (m << 4) + r + (hi << 3);
                votes[(((n << 8) + oo) << 12) + spb + (nt << 4) + lo] = acc[m][nt][r];
            }
}

// ---------------------------------------------------------------------------
// Dynamic routing, fused with final transpose.  One wave per spatial site
// (b, h, w); lane = output atom a.  Votes are read from HBM exactly once and
// held in registers (32 floats/lane) across all 3 routing iterations.
// Output goes through an LDS transpose so stores are contiguous 32B runs.
// ---------------------------------------------------------------------------
__global__ void __launch_bounds__(256)
routing_kernel(const float* __restrict__ votes,
               const float* __restrict__ bias,
               float* __restrict__ out) {
    __shared__ float ts[32][65];    // [a][o*8+site], padded vs bank conflicts

    int tid  = threadIdx.x;
    int lane = tid & 31;            // atom index a
    int site = tid >> 5;            // site within block (consecutive w)
    int s    = blockIdx.x * 8 + site;
    int b    = s >> 12;
    int sp   = s & 4095;            // h*64 + w

    // votes[b,i,h,w,o,a] = flat[(((b*4+i)*4096 + sp)*8 + o)*32 + a]
    float v[4][8];
    #pragma unroll
    for (int i = 0; i < 4; ++i) {
        size_t base = ((size_t)((b * 4 + i) * 4096 + sp)) * 256 + lane;
        #pragma unroll
        for (int o = 0; o < 8; ++o) v[i][o] = votes[base + ((size_t)o << 5)];
    }

    float bia[8];
    #pragma unroll
    for (int o = 0; o < 8; ++o) bia[o] = bias[(o << 5) + lane];

    float logit[4][8];
    #pragma unroll
    for (int i = 0; i < 4; ++i)
        #pragma unroll
        for (int o = 0; o < 8; ++o) logit[i][o] = 0.0f;

    float act[8];
    #pragma unroll
    for (int it = 0; it < 3; ++it) {
        // softmax over o per input capsule i (replicated across lanes)
        float route[4][8];
        #pragma unroll
        for (int i = 0; i < 4; ++i) {
            float mx = logit[i][0];
            #pragma unroll
            for (int o = 1; o < 8; ++o) mx = fmaxf(mx, logit[i][o]);
            float ssum = 0.0f;
            #pragma unroll
            for (int o = 0; o < 8; ++o) {
                route[i][o] = __expf(logit[i][o] - mx);
                ssum += route[i][o];
            }
            float inv = 1.0f / ssum;
            #pragma unroll
            for (int o = 0; o < 8; ++o) route[i][o] *= inv;
        }

        // preact + squash (wave32 butterfly over atom axis)
        #pragma unroll
        for (int o = 0; o < 8; ++o) {
            float p = bia[o];
            #pragma unroll
            for (int i = 0; i < 4; ++i) p = fmaf(route[i][o], v[i][o], p);
            float sq = p * p;
            #pragma unroll
            for (int off = 16; off > 0; off >>= 1)
                sq += __shfl_xor(sq, off, 32);
            act[o] = p * (sq / (1.0f + sq) * rsqrtf(sq + 1e-7f));
        }

        // agreement -> logits (skip on final iteration, unused)
        if (it < 2) {
            #pragma unroll
            for (int i = 0; i < 4; ++i)
                #pragma unroll
                for (int o = 0; o < 8; ++o) {
                    float ag = v[i][o] * act[o];
                    #pragma unroll
                    for (int off = 16; off > 0; off >>= 1)
                        ag += __shfl_xor(ag, off, 32);
                    logit[i][o] += ag;
                }
        }
    }

    // ---- transpose in LDS, then contiguous stores ----
    #pragma unroll
    for (int o = 0; o < 8; ++o) ts[lane][(o << 3) + site] = act[o];
    __syncthreads();

    int a  = tid >> 3;
    int oo = tid & 7;
    int spbase = (blockIdx.x * 8) & 4095;          // block covers 8 consecutive w
    size_t ob = ((size_t)(b * 32 + a) * 8 + oo) * 4096 + spbase;
    #pragma unroll
    for (int ss = 0; ss < 8; ++ss)
        out[ob + ss] = ts[a][(oo << 3) + ss];      // out (B, A_OUT, C_OUT, h, w)
}

// ---------------------------------------------------------------------------
extern "C" void kernel_launch(void* const* d_in, const int* in_sizes, int n_in,
                              void* d_out, int out_size, void* d_ws, size_t ws_size,
                              hipStream_t stream) {
    const float* x    = (const float*)d_in[0];   // (8,16,4,64,64)
    const float* W    = (const float*)d_in[1];   // (256,16,5,5)
    const float* bias = (const float*)d_in[2];   // (1,1,8,32)
    float* out = (float*)d_out;                  // (8,32,8,64,64)

    char* ws = (char*)d_ws;
    _Float16* inp   = (_Float16*)ws;                        //   4 MiB packed NHWC f16 input
    _Float16* wp    = (_Float16*)(ws + (size_t)(4u << 20)); // 208 KiB packed f16 weights
    float*    votes = (float*)(ws + (size_t)(8u << 20));    // 128 MiB fp32 conv output

    pack_input_kernel<<<8192, 256, 0, stream>>>(x, inp);          // 2,097,152 threads
    pack_w_kernel<<<416, 256, 0, stream>>>(W, wp);                //   106,496 threads
    conv_wmma_kernel<<<2048, 256, 0, stream>>>(inp, wp, votes);   // M=256 x N=64 tiles
    routing_kernel<<<4096, 256, 0, stream>>>(votes, bias, out);   //  32768 sites
}
